// Encoder_23527830847909
// MI455X (gfx1250) — matrix operand order repaired and checked by reading, not compiled, not run on hardware
//
#include <hip/hip_runtime.h>
#include <hip/hip_bf16.h>

typedef __attribute__((ext_vector_type(16))) _Float16 v16h;
typedef __attribute__((ext_vector_type(8)))  float    v8f;

#define EMB    128
#define HEADS  8
#define FFH    512
#define NLAY   6
#define NB     4
#define PICKN  250
#define NTOK   501          // 1 + 2*PICKN
#define NEGBIG (-30000.0f)  // effectively -inf after exp()
#define NF     0.25f        // 1/sqrt(DK=16)

__device__ __forceinline__ v8f wmma16(v16h a, v16h b, v8f c) {
  // D(f32 16x16) = A(f16 16x32) * B(f16 32x16) + C
  return __builtin_amdgcn_wmma_f32_16x16x32_f16(false, a, false, b, (short)0, c,
                                                false, false);
}

// ---------------------------------------------------------------------------
// GEMM: out[b, r, 0..Nout) = act[b, act_row0+r, 0..K) @ W[KxNout] + bias (opt ReLU)
// fp32 in/out, f16 WMMA with f32 accumulation. K, Nout compile-time -> the
// K-loop fully unrolls into back-to-back v_wmma ops with constant addressing.
// Grid: (ceil(Mb/16), Nout/16, nbatch); one wave per 16x16 output tile.
// ---------------------------------------------------------------------------
template<int K, int Nout>
__global__ void __launch_bounds__(32)
gemm_wmma(const float* __restrict__ act, int act_bstride, int act_row0,
          const float* __restrict__ W, const float* __restrict__ bias,
          float* __restrict__ out, int out_bstride, int Mb, int relu)
{
  const int lane = threadIdx.x;
  const int cm   = lane & 15;
  const int col  = blockIdx.y * 16 + cm;
  const int b    = blockIdx.z;
  const int kbA  = (lane < 16) ? 0 : 8;    // A-fragment K base (16-bit layout)
  const int kbB  = (lane < 16) ? 0 : 16;   // W row offset used only by the prefetch

  int r = blockIdx.x * 16 + cm;            // row this lane holds in A fragment
  if (r > Mb - 1) r = Mb - 1;              // clamp (duplicate rows, masked store)
  const float* arow = act + (size_t)b * act_bstride + (size_t)(act_row0 + r) * K;

  __builtin_prefetch(W + (size_t)kbB * Nout + col, 0, 1);  // global_prefetch_b8

  v8f acc = {};
#pragma unroll
  for (int kc = 0; kc < K; kc += 32) {
    v16h a, bf;
#pragma unroll
    for (int j = 0; j < 8; ++j) {
      a[j]     = (_Float16)arow[kc + kbA + j];
      a[8 + j] = (_Float16)arow[kc + 16 + kbA + j];
    }
#pragma unroll
    for (int e = 0; e < 16; ++e)
      bf[e] = (_Float16)W[(size_t)(kc + kbA + (e & 7) + ((e >> 3) << 4)) * Nout + col];
    acc = wmma16(a, bf, acc);
  }

  const float bv = bias ? bias[col] : 0.0f;
#pragma unroll
  for (int g = 0; g < 8; ++g) {
    const int m  = g + ((lane < 16) ? 0 : 8);
    const int gr = blockIdx.x * 16 + m;
    if (gr < Mb) {
      float v = acc[g] + bv;
      if (relu) v = fmaxf(v, 0.0f);
      out[(size_t)b * out_bstride + (size_t)gr * Nout + col] = v;
    }
  }
}

// ---------------------------------------------------------------------------
// Fused heterogeneous attention. One wave per (batch, head, 16-row tile).
// Per-row column set:
//   depot row (0):       score_n over N columns (k_n -> v_n)
//   pick row i (1..250): + q_p.(k_p->v_p)[250] + q_p.(k_d->v_d)[250] + paired (k_d[i],v_d[i])
//   deliv row j (251..): + q_d.(k_d->v_d)[250] + q_d.(k_p->v_p)[250] + paired (k_p[j],v_p[j])
// Scores tile in LDS (f16): cols [0,512)=node, [512,768)=k_p, [768,1024)=k_d,
// paired column kept separately.  Two-pass softmax in fp32, P@V via WMMA.
// ---------------------------------------------------------------------------
__global__ void __launch_bounds__(32)
attn_wmma(const float* __restrict__ qn, const float* __restrict__ kn, const float* __restrict__ vn,
          const float* __restrict__ qp, const float* __restrict__ kp, const float* __restrict__ vp,
          const float* __restrict__ qd, const float* __restrict__ kd, const float* __restrict__ vd,
          float* __restrict__ out)
{
  __shared__ _Float16 S[16 * 1024];     // 32 KB score / prob tile
  __shared__ float sp[16], ep[16], rsum[16], pmax[32], psum[32];

  const int lane = threadIdx.x;
  const int cm   = lane & 15;
  const int rt   = blockIdx.x;          // 16-row tile (0..31)
  const int h    = blockIdx.y;
  const int b    = blockIdx.z;
  const int ho   = h * 16;
  const int kbA  = (lane < 16) ? 0 : 8;
  const int kbB  = (lane < 16) ? 0 : 16;

  const float* qnb = qn + (size_t)b * NTOK * EMB;
  const float* knb = kn + (size_t)b * NTOK * EMB;
  const float* vnb = vn + (size_t)b * NTOK * EMB;
  const float* qpb = qp + (size_t)b * PICKN * EMB;
  const float* kpb = kp + (size_t)b * PICKN * EMB;
  const float* vpb = vp + (size_t)b * PICKN * EMB;
  const float* qdb = qd + (size_t)b * PICKN * EMB;
  const float* kdb = kd + (size_t)b * PICKN * EMB;
  const float* vdb = vd + (size_t)b * PICKN * EMB;

  const int grow  = rt * 16 + cm;                       // row this lane holds in A
  const int growc = (grow < NTOK) ? grow : (NTOK - 1);  // clamp for loads

  // A fragments: q_n and q_x (q_p for pick rows, q_d for delivery rows);
  // K = head dim (16), upper 16 K slots zero-padded.
  v16h aqn, aqx;
  {
    const float* qr = qnb + (size_t)growc * EMB + ho;
    const float* qx = nullptr;
    if (growc >= 1 && growc <= PICKN) qx = qpb + (size_t)(growc - 1) * EMB + ho;
    else if (growc > PICKN)           qx = qdb + (size_t)(growc - PICKN - 1) * EMB + ho;
#pragma unroll
    for (int j = 0; j < 8; ++j) {
      aqn[j]     = (lane < 16) ? (_Float16)qr[j] : (_Float16)0.0f;
      aqn[8 + j] = (lane < 16) ? (_Float16)qr[8 + j] : (_Float16)0.0f;
      aqx[j]     = (qx && lane < 16) ? (_Float16)qx[j] : (_Float16)0.0f;
      aqx[8 + j] = (qx && lane < 16) ? (_Float16)qx[8 + j] : (_Float16)0.0f;
    }
  }

  // ---- scores: node region (cols 0..511) ----
  for (int ct = 0; ct < 32; ++ct) {
    const int j  = ct * 16 + cm;
    const int jc = (j < NTOK) ? j : (NTOK - 1);
    v16h bf = {};                         // lanes>=16 hold K=16..31 -> zero pad
    if (lane < 16) {
      const float* kr = knb + (size_t)jc * EMB + ho;
#pragma unroll
      for (int e = 0; e < 16; ++e) bf[e] = (_Float16)kr[e];
    }
    v8f d = {};
    d = wmma16(aqn, bf, d);
#pragma unroll
    for (int g = 0; g < 8; ++g) {
      const int m = g + ((lane < 16) ? 0 : 8);
      const int c = ct * 16 + cm;
      S[m * 1024 + c] = (_Float16)((c < NTOK) ? d[g] * NF : NEGBIG);
    }
  }

  // ---- scores: extra regions (k_p at 512, k_d at 768); depot row masked ----
  for (int blk = 0; blk < 2; ++blk) {
    const float* kx = blk ? kdb : kpb;
    const int base  = 512 + blk * 256;
    for (int ct = 0; ct < 16; ++ct) {
      const int j  = ct * 16 + cm;
      const int jc = (j < PICKN) ? j : (PICKN - 1);
      v16h bf = {};
      if (lane < 16) {
        const float* kr = kx + (size_t)jc * EMB + ho;
#pragma unroll
        for (int e = 0; e < 16; ++e) bf[e] = (_Float16)kr[e];
      }
      v8f d = {};
      d = wmma16(aqx, bf, d);
#pragma unroll
      for (int g = 0; g < 8; ++g) {
        const int m  = g + ((lane < 16) ? 0 : 8);
        const int c  = ct * 16 + cm;
        const int gr = rt * 16 + m;
        const bool ok = (c < PICKN) && (gr >= 1) && (gr < NTOK);
        S[m * 1024 + base + c] = (_Float16)(ok ? d[g] * NF : NEGBIG);
      }
    }
  }

  // ---- paired-partner column (duplicated column per reference semantics) ----
  if (lane < 16) {
    const int gr = rt * 16 + lane;
    float s = NEGBIG;
    if (gr >= 1 && gr <= PICKN) {
      const float* qa = qpb + (size_t)(gr - 1) * EMB + ho;
      const float* ka = kdb + (size_t)(gr - 1) * EMB + ho;
      float d0 = 0.0f;
#pragma unroll
      for (int e = 0; e < 16; ++e) d0 += qa[e] * ka[e];
      s = d0 * NF;
    } else if (gr > PICKN && gr < NTOK) {
      const float* qa = qdb + (size_t)(gr - PICKN - 1) * EMB + ho;
      const float* ka = kpb + (size_t)(gr - PICKN - 1) * EMB + ho;
      float d0 = 0.0f;
#pragma unroll
      for (int e = 0; e < 16; ++e) d0 += qa[e] * ka[e];
      s = d0 * NF;
    }
    sp[lane] = s;
  }
  __syncthreads();

  // ---- softmax: each row scanned by 2 lanes (512 cols each), fp32 math ----
  const int srow  = lane & 15;
  const int rowb  = srow * 1024 + (lane >> 4) * 512;
  {
    float mx = -1e30f;
    for (int c = 0; c < 512; ++c) mx = fmaxf(mx, (float)S[rowb + c]);
    pmax[lane] = mx;
  }
  __syncthreads();
  const float mx = fmaxf(fmaxf(pmax[srow], pmax[srow + 16]), sp[srow]);
  {
    float sum = 0.0f;
    for (int c = 0; c < 512; ++c) {
      const float e_ = __expf((float)S[rowb + c] - mx);
      S[rowb + c] = (_Float16)e_;
      sum += e_;
    }
    psum[lane] = sum;
  }
  __syncthreads();
  if (lane < 16) {
    const float epair = __expf(sp[lane] - mx);
    ep[lane]   = epair;
    rsum[lane] = psum[lane] + psum[lane + 16] + epair;
  }
  __syncthreads();

  // ---- P @ V accumulation via WMMA (32-column chunks) ----
  v8f acc = {};
  for (int c = 0; c < 16; ++c) {              // node region -> v_n
    const int base = c * 32;
    v16h a, bf;
#pragma unroll
    for (int j = 0; j < 8; ++j) {
      a[j]     = S[cm * 1024 + base + kbA + j];
      a[8 + j] = S[cm * 1024 + base + 16 + kbA + j];
    }
#pragma unroll
    for (int e = 0; e < 16; ++e) {
      int j2 = base + kbA + (e & 7) + ((e >> 3) << 4);
      if (j2 > NTOK - 1) j2 = NTOK - 1;       // padded cols have weight 0
      bf[e] = (_Float16)vnb[(size_t)j2 * EMB + ho + cm];
    }
    acc = wmma16(a, bf, acc);
  }
  for (int blk = 0; blk < 2; ++blk) {         // extras: v_p then v_d
    const float* vx = blk ? vdb : vpb;
    const int sbase = 512 + blk * 256;
    for (int c = 0; c < 8; ++c) {
      const int base = c * 32;
      v16h a, bf;
#pragma unroll
      for (int j = 0; j < 8; ++j) {
        a[j]     = S[cm * 1024 + sbase + base + kbA + j];
        a[8 + j] = S[cm * 1024 + sbase + base + 16 + kbA + j];
      }
#pragma unroll
      for (int e = 0; e < 16; ++e) {
        int j2 = base + kbA + (e & 7) + ((e >> 3) << 4);
        if (j2 > PICKN - 1) j2 = PICKN - 1;
        bf[e] = (_Float16)vx[(size_t)j2 * EMB + ho + cm];
      }
      acc = wmma16(a, bf, acc);
    }
  }

  // ---- finalize: paired term + normalize + store (out is [B,N,H*DK]) ----
#pragma unroll
  for (int g = 0; g < 8; ++g) {
    const int m  = g + ((lane < 16) ? 0 : 8);
    const int gr = rt * 16 + m;
    if (gr < NTOK) {
      float vpr = 0.0f;
      if (gr >= 1 && gr <= PICKN) vpr = vdb[(size_t)(gr - 1) * EMB + ho + cm];
      else if (gr > PICKN)        vpr = vpb[(size_t)(gr - PICKN - 1) * EMB + ho + cm];
      const float val = (acc[g] + ep[m] * vpr) / rsum[m];
      out[((size_t)b * NTOK + gr) * EMB + ho + cm] = val;
    }
  }
}

// ---------------------------------------------------------------------------
// out = instance_norm(x + y) over the TOKEN axis (reference normalizes axis=1),
// per (batch, channel). One block per batch, one thread per channel.
// ---------------------------------------------------------------------------
__global__ void __launch_bounds__(EMB)
add_inorm(const float* __restrict__ x, const float* __restrict__ y,
          const float* __restrict__ g, const float* __restrict__ bb,
          float* __restrict__ out)
{
  const int b = blockIdx.x, c = threadIdx.x;
  const float* xb = x + (size_t)b * NTOK * EMB;
  const float* yb = y + (size_t)b * NTOK * EMB;
  float s = 0.0f, s2 = 0.0f;
  for (int n = 0; n < NTOK; ++n) {
    const float v = xb[n * EMB + c] + yb[n * EMB + c];
    s += v; s2 += v * v;
  }
  const float m   = s * (1.0f / NTOK);
  float var       = s2 * (1.0f / NTOK) - m * m;
  var             = fmaxf(var, 0.0f);
  const float r   = rsqrtf(var + 1e-5f);
  const float gc  = g[c], bc = bb[c];
  float* ob = out + (size_t)b * NTOK * EMB;
  for (int n = 0; n < NTOK; ++n) {
    const float v = xb[n * EMB + c] + yb[n * EMB + c];
    ob[n * EMB + c] = (v - m) * r * gc + bc;
  }
}

// ---------------------------------------------------------------------------
// Initial token embedding: e[b,t,c] = xy @ W(2x128) + bias, per node type.
// ---------------------------------------------------------------------------
__global__ void __launch_bounds__(256)
embed_kernel(const float* __restrict__ dxy, const float* __restrict__ pxy,
             const float* __restrict__ vxy,
             const float* __restrict__ Wdep, const float* __restrict__ bdep,
             const float* __restrict__ Wpick, const float* __restrict__ bpick,
             const float* __restrict__ Wdel, const float* __restrict__ bdel,
             float* __restrict__ e)
{
  const int idx = blockIdx.x * blockDim.x + threadIdx.x;
  if (idx >= NB * NTOK * EMB) return;
  const int c = idx & (EMB - 1);
  const int t = (idx >> 7) % NTOK;
  const int b = idx / (NTOK * EMB);
  float x, y; const float* W; const float* bi;
  if (t == 0)          { x = dxy[b * 2];  y = dxy[b * 2 + 1];  W = Wdep;  bi = bdep; }
  else if (t <= PICKN) { const int i = t - 1;
                         x = pxy[(b * PICKN + i) * 2]; y = pxy[(b * PICKN + i) * 2 + 1];
                         W = Wpick; bi = bpick; }
  else                 { const int i = t - PICKN - 1;
                         x = vxy[(b * PICKN + i) * 2]; y = vxy[(b * PICKN + i) * 2 + 1];
                         W = Wdel;  bi = bdel; }
  e[idx] = x * W[c] + y * W[EMB + c] + bi[c];
}

// ---------------------------------------------------------------------------
extern "C" void kernel_launch(void* const* d_in, const int* in_sizes, int n_in,
                              void* d_out, int out_size, void* d_ws, size_t ws_size,
                              hipStream_t stream) {
  (void)in_sizes; (void)n_in; (void)out_size; (void)ws_size;

  // JAX pytree (sorted-key) flattening order.
  const float* dxy   = (const float*)d_in[0];
  const float* pxy   = (const float*)d_in[1];
  const float* vxy   = (const float*)d_in[2];
  const float* Wdel  = (const float*)d_in[3];
  const float* Wdep  = (const float*)d_in[4];
  const float* Wpick = (const float*)d_in[5];
  const float* bdel  = (const float*)d_in[6];
  const float* bdep  = (const float*)d_in[7];
  const float* bpick = (const float*)d_in[8];

  struct Lay { const float *W1,*W2,*Wc,*Wk_d,*Wk_n,*Wk_p,*Wq_d,*Wq_n,*Wq_p,
                           *Wv_d,*Wv_n,*Wv_p,*b1,*b2,*bW1,*bW2,*bc,*g1,*g2; };
  Lay L[NLAY];
  int p = 9;
  for (int l = 0; l < NLAY; ++l) {
    const float** f = (const float**)&L[l];
    for (int k = 0; k < 19; ++k) f[k] = (const float*)d_in[p++];
  }

  // Workspace carve-up (floats).
  float* ws = (float*)d_ws;
  const size_t SB = (size_t)NB * NTOK * EMB;   // 256512
  const size_t SP = (size_t)NB * PICKN * EMB;  // 128000
  float* e     = ws; ws += SB;
  float* qn    = ws; ws += SB;
  float* kn    = ws; ws += SB;
  float* vn    = ws; ws += SB;
  float* qpb   = ws; ws += SP;
  float* kpb   = ws; ws += SP;
  float* vpb   = ws; ws += SP;
  float* qdb   = ws; ws += SP;
  float* kdb   = ws; ws += SP;
  float* vdb   = ws; ws += SP;
  float* ao    = ws; ws += SB;
  float* ctmp  = ws; ws += SB;
  float* h1    = ws; ws += SB;
  float* ffb   = ws; ws += (size_t)NB * NTOK * FFH;
  float* fftmp = ws; ws += SB;

  embed_kernel<<<(NB * NTOK * EMB + 255) / 256, 256, 0, stream>>>(
      dxy, pxy, vxy, Wdep, bdep, Wpick, bpick, Wdel, bdel, e);

  // GEMM dispatch over the three static shapes.
  auto gemm128x128 = [&](const float* act, int abst, int arow0, const float* W,
                         const float* bias, float* out, int obst, int Mb, int relu) {
    dim3 grid((Mb + 15) / 16, EMB / 16, NB);
    gemm_wmma<EMB, EMB><<<grid, 32, 0, stream>>>(act, abst, arow0, W, bias, out,
                                                 obst, Mb, relu);
  };

  for (int l = 0; l < NLAY; ++l) {
    const Lay& P = L[l];
    // QKV projections (full set, pick slice, delivery slice)
    gemm128x128(e, NTOK * EMB, 0,         P.Wq_n, nullptr, qn,  NTOK * EMB,  NTOK,  0);
    gemm128x128(e, NTOK * EMB, 0,         P.Wk_n, nullptr, kn,  NTOK * EMB,  NTOK,  0);
    gemm128x128(e, NTOK * EMB, 0,         P.Wv_n, nullptr, vn,  NTOK * EMB,  NTOK,  0);
    gemm128x128(e, NTOK * EMB, 1,         P.Wq_p, nullptr, qpb, PICKN * EMB, PICKN, 0);
    gemm128x128(e, NTOK * EMB, 1,         P.Wk_p, nullptr, kpb, PICKN * EMB, PICKN, 0);
    gemm128x128(e, NTOK * EMB, 1,         P.Wv_p, nullptr, vpb, PICKN * EMB, PICKN, 0);
    gemm128x128(e, NTOK * EMB, 1 + PICKN, P.Wq_d, nullptr, qdb, PICKN * EMB, PICKN, 0);
    gemm128x128(e, NTOK * EMB, 1 + PICKN, P.Wk_d, nullptr, kdb, PICKN * EMB, PICKN, 0);
    gemm128x128(e, NTOK * EMB, 1 + PICKN, P.Wv_d, nullptr, vdb, PICKN * EMB, PICKN, 0);
    // fused heterogeneous attention
    attn_wmma<<<dim3((NTOK + 15) / 16, HEADS, NB), 32, 0, stream>>>(
        qn, kn, vn, qpb, kpb, vpb, qdb, kdb, vdb, ao);
    // output projection + residual + inorm
    gemm128x128(ao, NTOK * EMB, 0, P.Wc, P.bc, ctmp, NTOK * EMB, NTOK, 0);
    add_inorm<<<NB, EMB, 0, stream>>>(e, ctmp, P.g1, P.b1, h1);
    // FFN + residual + inorm
    {
      dim3 g1((NTOK + 15) / 16, FFH / 16, NB);
      gemm_wmma<EMB, FFH><<<g1, 32, 0, stream>>>(h1, NTOK * EMB, 0, P.W1, P.bW1,
                                                 ffb, NTOK * FFH, NTOK, 1);
      dim3 g2((NTOK + 15) / 16, EMB / 16, NB);
      gemm_wmma<FFH, EMB><<<g2, 32, 0, stream>>>(ffb, NTOK * FFH, 0, P.W2, P.bW2,
                                                 fftmp, NTOK * EMB, NTOK, 0);
    }
    float* dst = (l == NLAY - 1) ? (float*)d_out : e;
    add_inorm<<<NB, EMB, 0, stream>>>(h1, fftmp, P.g2, P.b2, dst);
  }
}
